// Encoder_85375359910465
// MI455X (gfx1250) — matrix operand order repaired and checked by reading, not compiled, not run on hardware
//
#include <hip/hip_runtime.h>
#include <hip/hip_bf16.h>

// ---------------------------------------------------------------------------
// CDNA5 / gfx1250 implementation. All dense contractions use
// v_wmma_f32_16x16x32_bf16 (bf16 in, f32 accumulate). wave32 everywhere.
// Big GEMMs are register-blocked 64x64 per wave (4x4 WMMA tiles);
// __launch_bounds__(128,1) gives the allocator the full VGPR budget so the
// 128 accumulator VGPRs stay resident (no scratch spills).
// ---------------------------------------------------------------------------

typedef __attribute__((ext_vector_type(16))) __bf16 v16bf;
typedef __attribute__((ext_vector_type(8)))  __bf16 v8bf;
typedef __attribute__((ext_vector_type(8)))  float  v8f;

#define BF16(x) ((__bf16)(x))

__device__ __forceinline__ v16bf load_a_frag(const __bf16* row, int k0, int khalf)
{
    v8bf lo = *(const v8bf*)(row + k0 + khalf * 16);
    v8bf hi = *(const v8bf*)(row + k0 + khalf * 16 + 8);
    return __builtin_shufflevector(lo, hi,
           0,1,2,3,4,5,6,7,8,9,10,11,12,13,14,15);
}

// ---------------------------------------------------------------------------
// Register-blocked GEMM: C[m,n] = sum_k A[m,k]*Bt[n,k] (+bias).
// A row-major [M,K] (bf16), Bt row-major [N,K] (bf16).
// Block = (32,4): 4 waves in a 2x2 grid; each wave computes 64x64 of C
// (4x4 WMMA tiles) -> block tile 128M x 128N. K % 32 == 0.
// ---------------------------------------------------------------------------
__global__ void __launch_bounds__(128, 1)
gemm_bt_wmma64(const __bf16* __restrict__ A,
               const __bf16* __restrict__ Bt,
               const float*  __restrict__ bias,
               float*  __restrict__ Cf,
               __bf16* __restrict__ Cb,
               int M, int N, int K,
               int lda, int ldb, int ldc)
{
    int lane  = threadIdx.x;
    int l15   = lane & 15;
    int khalf = lane >> 4;
    int wm = threadIdx.y & 1;
    int wn = threadIdx.y >> 1;
    int m0 = blockIdx.x * 128 + wm * 64;
    int n0 = blockIdx.y * 128 + wn * 64;

    const __bf16* Arow[4];
    const __bf16* Brow[4];
    #pragma unroll
    for (int i = 0; i < 4; ++i) {
        int mrow = m0 + i * 16 + l15; if (mrow >= M) mrow = M - 1;
        Arow[i] = A + (long long)mrow * lda;
    }
    #pragma unroll
    for (int j = 0; j < 4; ++j) {
        int nrow = n0 + j * 16 + l15; if (nrow >= N) nrow = N - 1;
        Brow[j] = Bt + (long long)nrow * ldb;
    }

    v8f acc[4][4] = {};
    for (int k0 = 0; k0 < K; k0 += 32) {
        v16bf b[4];
        #pragma unroll
        for (int j = 0; j < 4; ++j)
            b[j] = *(const v16bf*)(Brow[j] + k0 + khalf * 16);
        #pragma unroll
        for (int i = 0; i < 4; ++i) {
            v16bf a = load_a_frag(Arow[i], k0, khalf);
            #pragma unroll
            for (int j = 0; j < 4; ++j)
                acc[i][j] = __builtin_amdgcn_wmma_f32_16x16x32_bf16(
                    false, a, false, b[j], (short)0, acc[i][j], false, false);
        }
    }

    #pragma unroll
    for (int j = 0; j < 4; ++j) {
        int ncol = n0 + j * 16 + l15;
        if (ncol >= N) continue;
        float bv = bias ? bias[ncol] : 0.0f;
        #pragma unroll
        for (int i = 0; i < 4; ++i) {
            #pragma unroll
            for (int r = 0; r < 8; ++r) {
                int m = m0 + i * 16 + r + 8 * khalf;
                if (m < M) {
                    float v = acc[i][j][r] + bv;
                    long long off = (long long)m * ldc + ncol;
                    if (Cf) Cf[off] = v;
                    if (Cb) Cb[off] = BF16(v);
                }
            }
        }
    }
}

// ---------------------------------------------------------------------------
// Register-blocked implicit-im2col conv GEMM (convs 2..6).
// act_in [B,T_in,cin] bf16, wt [cout, k*cin] bf16 (kk = dk*cin+ci),
// out [B,T_out,cout] bf16 (+bias). cin power of two, K = k*cin % 32 == 0.
// Same 64x64-per-wave blocking as gemm_bt_wmma64.
// ---------------------------------------------------------------------------
__global__ void __launch_bounds__(128, 1)
conv_bt_wmma64(const __bf16* __restrict__ act_in,
               const __bf16* __restrict__ wt,
               const float*  __restrict__ bias,
               __bf16* __restrict__ act_out,
               int M, int T_out, int T_in,
               int log2cin, int stride, int cout, int K)
{
    int lane  = threadIdx.x;
    int l15   = lane & 15;
    int khalf = lane >> 4;
    int wm = threadIdx.y & 1;
    int wn = threadIdx.y >> 1;
    int m0 = blockIdx.x * 128 + wm * 64;
    int n0 = blockIdx.y * 128 + wn * 64;
    int cinm1 = (1 << log2cin) - 1;

    const __bf16* Abase[4];
    #pragma unroll
    for (int i = 0; i < 4; ++i) {
        int mrow = m0 + i * 16 + l15; if (mrow >= M) mrow = M - 1;
        int b = mrow / T_out;
        int t = mrow - b * T_out;
        Abase[i] = act_in + (((long long)b * T_in + (long long)t * stride) << log2cin);
    }
    const __bf16* Brow[4];
    #pragma unroll
    for (int j = 0; j < 4; ++j) {
        int nrow = n0 + j * 16 + l15; if (nrow >= cout) nrow = cout - 1;
        Brow[j] = wt + (long long)nrow * K;
    }

    v8f acc[4][4] = {};
    for (int k0 = 0; k0 < K; k0 += 32) {
        int kk0 = k0 + khalf * 16;
        int kk1 = k0 + khalf * 16 + 8;
        long long off0 = ((long long)(kk0 >> log2cin) << log2cin) + (kk0 & cinm1);
        long long off1 = ((long long)(kk1 >> log2cin) << log2cin) + (kk1 & cinm1);
        v16bf b[4];
        #pragma unroll
        for (int j = 0; j < 4; ++j)
            b[j] = *(const v16bf*)(Brow[j] + k0 + khalf * 16);
        #pragma unroll
        for (int i = 0; i < 4; ++i) {
            v8bf lo = *(const v8bf*)(Abase[i] + off0);
            v8bf hi = *(const v8bf*)(Abase[i] + off1);
            v16bf a = __builtin_shufflevector(lo, hi,
                      0,1,2,3,4,5,6,7,8,9,10,11,12,13,14,15);
            #pragma unroll
            for (int j = 0; j < 4; ++j)
                acc[i][j] = __builtin_amdgcn_wmma_f32_16x16x32_bf16(
                    false, a, false, b[j], (short)0, acc[i][j], false, false);
        }
    }

    #pragma unroll
    for (int j = 0; j < 4; ++j) {
        int ncol = n0 + j * 16 + l15;
        if (ncol >= cout) continue;
        float bv = bias[ncol];
        #pragma unroll
        for (int i = 0; i < 4; ++i) {
            #pragma unroll
            for (int r = 0; r < 8; ++r) {
                int m = m0 + i * 16 + r + 8 * khalf;
                if (m < M)
                    act_out[(long long)m * cout + ncol] = BF16(acc[i][j][r] + bv);
            }
        }
    }
}

// ---------------------------------------------------------------------------
// Small batched GEMM (attention 57x57 / 57x64 tiles). One 16x16 tile per
// wave, block (32,4). Batch offset: off = (z>>3)*Xo + (z&7)*Xi.
// ---------------------------------------------------------------------------
__global__ void gemm_bt_small(const __bf16* __restrict__ A,
                              const __bf16* __restrict__ Bt,
                              float*  __restrict__ Cf,
                              __bf16* __restrict__ Cb,
                              int M, int N, int K,
                              int lda, int ldb, int ldc,
                              long long Ao, long long Ai,
                              long long Bo, long long Bi,
                              long long Co, long long Ci)
{
    int z = blockIdx.z;
    const __bf16* Ab = A  + (long long)(z >> 3) * Ao + (long long)(z & 7) * Ai;
    const __bf16* Bb = Bt + (long long)(z >> 3) * Bo + (long long)(z & 7) * Bi;
    long long cbase  =      (long long)(z >> 3) * Co + (long long)(z & 7) * Ci;

    int lane  = threadIdx.x;
    int l15   = lane & 15;
    int khalf = lane >> 4;
    int m0 = blockIdx.x * 16;
    int n0 = (blockIdx.y * 4 + threadIdx.y) * 16;

    int mrow = m0 + l15; if (mrow >= M) mrow = M - 1;
    int nrow = n0 + l15; if (nrow >= N) nrow = N - 1;

    const __bf16* Arow = Ab + (long long)mrow * lda;
    const __bf16* Brow = Bb + (long long)nrow * ldb;

    v8f acc = {};
    for (int k0 = 0; k0 < K; k0 += 32) {
        v16bf a = load_a_frag(Arow, k0, khalf);
        v16bf b = *(const v16bf*)(Brow + k0 + khalf * 16);
        acc = __builtin_amdgcn_wmma_f32_16x16x32_bf16(
                  false, a, false, b, (short)0, acc, false, false);
    }

    int ncol = n0 + l15;
    if (ncol < N) {
        #pragma unroll
        for (int r = 0; r < 8; ++r) {
            int m = m0 + r + 8 * khalf;
            if (m < M) {
                long long off = cbase + (long long)m * ldc + ncol;
                if (Cf) Cf[off] = acc[r];
                if (Cb) Cb[off] = BF16(acc[r]);
            }
        }
    }
}

// ---------------------------------------------------------------------------
// Conv1: cin=105, k=10, stride=3 (K=1050 not 32-aligned) -> direct f32,
// weight row cached in LDS. x [64,105,5500] f32 -> act1 [64,1831,64] bf16.
// ---------------------------------------------------------------------------
__global__ void conv1_direct(const float* __restrict__ x,
                             const float* __restrict__ w1,
                             const float* __restrict__ b1,
                             __bf16* __restrict__ act1)
{
    __shared__ float wrow[1050];
    int co = blockIdx.y;
    int b  = blockIdx.z;
    for (int i = threadIdx.x; i < 1050; i += blockDim.x)
        wrow[i] = w1[co * 1050 + i];
    __syncthreads();
    int t = blockIdx.x * blockDim.x + threadIdx.x;
    if (t >= 1831) return;
    const float* xb = x + (long long)b * 105 * 5500 + t * 3;
    float acc = b1[co];
    for (int ci = 0; ci < 105; ++ci) {
        const float* xr = xb + ci * 5500;
        const float* wr = wrow + ci * 10;
        #pragma unroll
        for (int dk = 0; dk < 10; ++dk)
            acc += xr[dk] * wr[dk];
    }
    act1[((long long)b * 1831 + t) * 64 + co] = BF16(acc);
}

// ----------------------------- weight prep ---------------------------------
__global__ void prep_conv_w(const float* __restrict__ w, __bf16* __restrict__ wt,
                            int cout, int cin, int ksz)
{
    int K = cin * ksz;
    int idx = blockIdx.x * blockDim.x + threadIdx.x;
    if (idx >= cout * K) return;
    int co = idx / K, kk = idx - co * K;
    int dk = kk / cin, ci = kk - dk * cin;
    wt[idx] = BF16(w[(co * cin + ci) * ksz + dk]);
}

// wq [8,512,64] f32 -> wqt [(h*64+j)][d] bf16  (N=512 rows, K=512)
__global__ void prep_qkv_w(const float* __restrict__ w, __bf16* __restrict__ wt)
{
    int idx = blockIdx.x * blockDim.x + threadIdx.x;
    if (idx >= 512 * 512) return;
    int n = idx >> 9, d = idx & 511;
    int h = n >> 6,  j = n & 63;
    wt[idx] = BF16(w[((h << 9) + d) * 64 + j]);
}

// wo [512,512] (din,dout) -> wot[n][k] = wo[k][n] bf16
__global__ void prep_wo(const float* __restrict__ w, __bf16* __restrict__ wt)
{
    int idx = blockIdx.x * blockDim.x + threadIdx.x;
    if (idx >= 512 * 512) return;
    int n = idx >> 9, k = idx & 511;
    wt[idx] = BF16(w[(k << 9) + n]);
}

__global__ void prep_cb(const float* __restrict__ cb, __bf16* __restrict__ out)
{
    int idx = blockIdx.x * blockDim.x + threadIdx.x;
    if (idx >= 2048 * 512) return;
    out[idx] = BF16(cb[idx]);
}

__global__ void cb_norm_kernel(const float* __restrict__ cb, float* __restrict__ cnorm)
{
    __shared__ float s[128];
    int n = blockIdx.x;
    float acc = 0.f;
    for (int i = threadIdx.x; i < 512; i += 128) {
        float v = cb[(long long)n * 512 + i];
        acc += v * v;
    }
    s[threadIdx.x] = acc; __syncthreads();
    for (int st = 64; st > 0; st >>= 1) {
        if (threadIdx.x < st) s[threadIdx.x] += s[threadIdx.x + st];
        __syncthreads();
    }
    if (threadIdx.x == 0) cnorm[n] = s[0];
}

// V [B,57,H*64] bf16 -> Vt [B,H,64(j),64(m)] bf16, rows m>=57 zero-padded.
__global__ void transpose_v(const __bf16* __restrict__ V, __bf16* __restrict__ Vt)
{
    int idx = blockIdx.x * blockDim.x + threadIdx.x;
    if (idx >= 64 * 8 * 64 * 64) return;
    int m = idx & 63;
    int j = (idx >> 6) & 63;
    int h = (idx >> 12) & 7;
    int b = idx >> 15;
    __bf16 v = (m < 57) ? V[(((long long)b * 57 + m) << 9) + (h << 6) + j] : BF16(0.f);
    Vt[idx] = v;
}

// One wave per (b,h,l); scores [B,H,57,57] f32 -> attn_p [B,H,64,64] bf16,
// zero-padded (cols/rows >= 57). Scale 1/sqrt(64)=0.125 fused.
__global__ void softmax_kernel(const float* __restrict__ scores,
                               __bf16* __restrict__ attn_p)
{
    int row = blockIdx.x;
    int l  = row & 63;
    int bh = row >> 6;
    int lane = threadIdx.x;
    __bf16* out = attn_p + (((long long)bh << 6) + l) * 64;
    if (l >= 57) { out[lane] = BF16(0.f); out[lane + 32] = BF16(0.f); return; }
    const float* s = scores + ((long long)bh * 57 + l) * 57;
    float v0 = (lane      < 57) ? s[lane]      * 0.125f : -3.0e38f;
    float v1 = (lane + 32 < 57) ? s[lane + 32] * 0.125f : -3.0e38f;
    float mx = fmaxf(v0, v1);
    for (int off = 16; off > 0; off >>= 1) mx = fmaxf(mx, __shfl_xor(mx, off, 32));
    float e0 = (lane      < 57) ? __expf(v0 - mx) : 0.f;
    float e1 = (lane + 32 < 57) ? __expf(v1 - mx) : 0.f;
    float sum = e0 + e1;
    for (int off = 16; off > 0; off >>= 1) sum += __shfl_xor(sum, off, 32);
    float inv = 1.f / sum;
    out[lane]      = BF16(e0 * inv);
    out[lane + 32] = BF16(e1 * inv);
}

// argmin_n (||c_n||^2 - 2*dot[r,n]); ties -> smallest index (matches argmin).
__global__ void argmin_kernel(const float* __restrict__ dot,
                              const float* __restrict__ cnorm,
                              int* __restrict__ idx_out,
                              int* __restrict__ idx_ws)
{
    __shared__ float sv[256];
    __shared__ int   si[256];
    int r = blockIdx.x;
    const float* d = dot + (long long)r * 2048;
    float best = 3.0e38f; int bi = 0;
    for (int n = threadIdx.x; n < 2048; n += 256) {
        float v = cnorm[n] - 2.f * d[n];
        if (v < best) { best = v; bi = n; }
    }
    sv[threadIdx.x] = best; si[threadIdx.x] = bi;
    __syncthreads();
    for (int st = 128; st > 0; st >>= 1) {
        if (threadIdx.x < st) {
            float ov = sv[threadIdx.x + st]; int oi = si[threadIdx.x + st];
            if (ov < sv[threadIdx.x] ||
                (ov == sv[threadIdx.x] && oi < si[threadIdx.x])) {
                sv[threadIdx.x] = ov; si[threadIdx.x] = oi;
            }
        }
        __syncthreads();
    }
    if (threadIdx.x == 0) { idx_out[r] = si[0]; idx_ws[r] = si[0]; }
}

// z_q_st == codebook[idx] exactly (straight-through forward value).
__global__ void gather_zq(const float* __restrict__ cb,
                          const int* __restrict__ idx,
                          float* __restrict__ zq)
{
    long long i = (long long)blockIdx.x * blockDim.x + threadIdx.x;
    if (i >= 3648LL * 512) return;
    int r = (int)(i >> 9), c = (int)(i & 511);
    zq[i] = cb[((long long)idx[r] << 9) + c];
}

__global__ void loss_partial(const float* __restrict__ zq,
                             const float* __restrict__ attn,
                             float* __restrict__ part)
{
    __shared__ float s[256];
    const long long n = 3648LL * 512;
    float acc = 0.f;
    for (long long i = (long long)blockIdx.x * 256 + threadIdx.x; i < n;
         i += 256LL * gridDim.x) {
        float d = zq[i] - attn[i];
        acc += d * d;
    }
    s[threadIdx.x] = acc; __syncthreads();
    for (int st = 128; st > 0; st >>= 1) {
        if (threadIdx.x < st) s[threadIdx.x] += s[threadIdx.x + st];
        __syncthreads();
    }
    if (threadIdx.x == 0) part[blockIdx.x] = s[0];
}

__global__ void loss_final(const float* __restrict__ part, int nparts,
                           float* __restrict__ out)
{
    __shared__ float s[256];
    float acc = 0.f;
    for (int i = threadIdx.x; i < nparts; i += 256) acc += part[i];
    s[threadIdx.x] = acc; __syncthreads();
    for (int st = 128; st > 0; st >>= 1) {
        if (threadIdx.x < st) s[threadIdx.x] += s[threadIdx.x + st];
        __syncthreads();
    }
    if (threadIdx.x == 0)
        out[0] = 1.2f * s[0] / (3648.0f * 512.0f);   // (1+beta)*mean
}

// ---------------------------------------------------------------------------
extern "C" void kernel_launch(void* const* d_in, const int* in_sizes, int n_in,
                              void* d_out, int out_size, void* d_ws, size_t ws_size,
                              hipStream_t stream)
{
    (void)in_sizes; (void)n_in; (void)out_size; (void)ws_size;
    const float* x  = (const float*)d_in[0];
    const float* cw[7]; const float* cbias[7];
    for (int i = 1; i <= 6; ++i) { cw[i] = (const float*)d_in[2*i-1]; cbias[i] = (const float*)d_in[2*i]; }
    const float* wq = (const float*)d_in[13];
    const float* wk = (const float*)d_in[14];
    const float* wv = (const float*)d_in[15];
    const float* wo = (const float*)d_in[16];
    const float* bo = (const float*)d_in[17];
    const float* codebook = (const float*)d_in[18];

    char* wsb = (char*)d_ws;
    size_t off = 0;
    auto alloc = [&](size_t bytes) -> void* {
        void* p = wsb + off;
        off = (off + bytes + 255) & ~(size_t)255;
        return p;
    };

    // activations [B,T,C] bf16
    __bf16* act1 = (__bf16*)alloc(64LL*1831*64*2);
    __bf16* act2 = (__bf16*)alloc(64LL*915*128*2);
    __bf16* act3 = (__bf16*)alloc(64LL*457*256*2);
    __bf16* act4 = (__bf16*)alloc(64LL*228*512*2);
    __bf16* act5 = (__bf16*)alloc(64LL*114*512*2);
    __bf16* act6 = (__bf16*)alloc(64LL*57*512*2);
    // weights (bf16, [N,K])
    __bf16* wt2 = (__bf16*)alloc(128LL*192*2);
    __bf16* wt3 = (__bf16*)alloc(256LL*384*2);
    __bf16* wt4 = (__bf16*)alloc(512LL*768*2);
    __bf16* wt5 = (__bf16*)alloc(512LL*1024*2);
    __bf16* wt6 = (__bf16*)alloc(512LL*1024*2);
    __bf16* wqt = (__bf16*)alloc(512LL*512*2);
    __bf16* wkt = (__bf16*)alloc(512LL*512*2);
    __bf16* wvt = (__bf16*)alloc(512LL*512*2);
    __bf16* wot = (__bf16*)alloc(512LL*512*2);
    // attention buffers
    __bf16* Q   = (__bf16*)alloc(3648LL*512*2);
    __bf16* Km  = (__bf16*)alloc(3648LL*512*2);
    __bf16* V   = (__bf16*)alloc(3648LL*512*2);
    __bf16* Vt  = (__bf16*)alloc(64LL*8*64*64*2);
    __bf16* attn_p = (__bf16*)alloc(64LL*8*64*64*2);
    __bf16* ctx = (__bf16*)alloc(3648LL*512*2);
    // VQ buffers
    __bf16* cbb   = (__bf16*)alloc(2048LL*512*2);
    float*  cnorm = (float*)alloc(2048*4);
    int*    idxws = (int*)alloc(3648*4);
    float*  part  = (float*)alloc(1024*4);
    // aliased scratch (source buffers are consumed before these are written):
    float*  dot    = (float*)act1;          // 29.9 MB over act1+act2
    float*  attnf  = (float*)act3;          // 7.5 MB
    __bf16* attnb  = (__bf16*)act4;         // 3.7 MB
    float*  scores = (float*)act5;          // 6.65 MB

    float* out_zq   = (float*)d_out;                       // [3648,512]
    float* out_loss = (float*)d_out + 1867776;             // scalar
    int*   out_idx  = (int*)((float*)d_out + 1867777);     // [3648] int32

    dim3 wblk(32, 4);

    // --- weight / codebook prep ---
    prep_conv_w<<<(128*192 +255)/256, 256, 0, stream>>>(cw[2], wt2, 128, 64, 3);
    prep_conv_w<<<(256*384 +255)/256, 256, 0, stream>>>(cw[3], wt3, 256, 128, 3);
    prep_conv_w<<<(512*768 +255)/256, 256, 0, stream>>>(cw[4], wt4, 512, 256, 3);
    prep_conv_w<<<(512*1024+255)/256, 256, 0, stream>>>(cw[5], wt5, 512, 512, 2);
    prep_conv_w<<<(512*1024+255)/256, 256, 0, stream>>>(cw[6], wt6, 512, 512, 2);
    prep_qkv_w<<<1024, 256, 0, stream>>>(wq, wqt);
    prep_qkv_w<<<1024, 256, 0, stream>>>(wk, wkt);
    prep_qkv_w<<<1024, 256, 0, stream>>>(wv, wvt);
    prep_wo<<<1024, 256, 0, stream>>>(wo, wot);
    prep_cb<<<4096, 256, 0, stream>>>(codebook, cbb);
    cb_norm_kernel<<<2048, 128, 0, stream>>>(codebook, cnorm);

    // --- conv stack ---
    conv1_direct<<<dim3(8, 64, 64), 256, 0, stream>>>(x, cw[1], cbias[1], act1);
    // grid.x = ceil(M/128), grid.y = ceil(cout/128)
    conv_bt_wmma64<<<dim3(458, 1, 1), wblk, 0, stream>>>(act1, wt2, cbias[2], act2,
        58560, 915, 1831, 6, 2, 128, 192);
    conv_bt_wmma64<<<dim3(229, 2, 1), wblk, 0, stream>>>(act2, wt3, cbias[3], act3,
        29248, 457, 915, 7, 2, 256, 384);
    conv_bt_wmma64<<<dim3(114, 4, 1), wblk, 0, stream>>>(act3, wt4, cbias[4], act4,
        14592, 228, 457, 8, 2, 512, 768);
    conv_bt_wmma64<<<dim3(57, 4, 1), wblk, 0, stream>>>(act4, wt5, cbias[5], act5,
        7296, 114, 228, 9, 2, 512, 1024);
    conv_bt_wmma64<<<dim3(29, 4, 1), wblk, 0, stream>>>(act5, wt6, cbias[6], act6,
        3648, 57, 114, 9, 2, 512, 1024);

    // --- attention ---
    // Q/K/V projections: [3648,512] x [512,512]
    gemm_bt_wmma64<<<dim3(29, 4, 1), wblk, 0, stream>>>(act6, wqt, nullptr, nullptr, Q,
        3648, 512, 512, 512, 512, 512);
    gemm_bt_wmma64<<<dim3(29, 4, 1), wblk, 0, stream>>>(act6, wkt, nullptr, nullptr, Km,
        3648, 512, 512, 512, 512, 512);
    gemm_bt_wmma64<<<dim3(29, 4, 1), wblk, 0, stream>>>(act6, wvt, nullptr, nullptr, V,
        3648, 512, 512, 512, 512, 512);
    // scores[b,h] = Q[b,:,h*64:] * K[b,:,h*64:]^T  (512 batches, 57x57, K=64)
    gemm_bt_small<<<dim3(4, 1, 512), wblk, 0, stream>>>(Q, Km, scores, nullptr,
        57, 57, 64, 512, 512, 57,
        57LL*512, 64, 57LL*512, 64, 8LL*57*57, 57LL*57);
    softmax_kernel<<<64*8*64, 32, 0, stream>>>(scores, attn_p);
    transpose_v<<<(64*8*64*64+255)/256, 256, 0, stream>>>(V, Vt);
    // ctx[b,h] = attn_p[b,h] (64x64) * Vt[b,h]^T (64x64)
    gemm_bt_small<<<dim3(4, 1, 512), wblk, 0, stream>>>(attn_p, Vt, nullptr, ctx,
        57, 64, 64, 64, 64, 512,
        8LL*4096, 4096, 8LL*4096, 4096, 57LL*512, 64);
    // output projection (+bo), write f32 (for loss) and bf16 (for VQ GEMM)
    gemm_bt_wmma64<<<dim3(29, 4, 1), wblk, 0, stream>>>(ctx, wot, bo, attnf, attnb,
        3648, 512, 512, 512, 512, 512);

    // --- VQ ---
    // dot = attn * codebook^T : [3648,512] x [2048,512]^T
    gemm_bt_wmma64<<<dim3(29, 16, 1), wblk, 0, stream>>>(attnb, cbb, nullptr, dot, nullptr,
        3648, 2048, 512, 512, 512, 2048);
    argmin_kernel<<<3648, 256, 0, stream>>>(dot, cnorm, out_idx, idxws);
    gather_zq<<<(3648*512+255)/256, 256, 0, stream>>>(codebook, idxws, out_zq);
    loss_partial<<<1024, 256, 0, stream>>>(out_zq, attnf, part);
    loss_final<<<1, 256, 0, stream>>>(part, 1024, out_loss);
}
